// GridImplicit_16217796510070
// MI455X (gfx1250) — compile-verified, matrix operand order checked by reading, not run on hardware
//
#include <hip/hip_runtime.h>

typedef __attribute__((ext_vector_type(16))) _Float16 v16h;
typedef __attribute__((ext_vector_type(8)))  _Float16 v8h;
typedef __attribute__((ext_vector_type(8)))  float    v8f;

constexpr int NLEV = 5;
constexpr int HID  = 128;

// ---------------- LDS layout (units: halves unless noted) ----------------
constexpr int W0_H   = 4 * 2 * 128 * 8;            // K=64  (2 ksteps) x N=128
constexpr int W1_H   = 4 * 4 * 128 * 8;            // K=128 (4 ksteps) x N=128
constexpr int W2_H   = 4 * 4 * 8;                  // col-0 only: 128 halves
constexpr int LVL_H  = W0_H + W1_H + W2_H;         // 24704 halves / level
constexpr int WTOT_H = NLEV * LVL_H;               // 123520 halves
constexpr int ACT_ROW_H  = 136;                    // 128 + 8 pad (272B rows)
constexpr int ACT_TILE_H = 16 * ACT_ROW_H;
constexpr int ACT_WAVE_H = 2 * ACT_TILE_H;         // two 16-pt tiles per wave
constexpr int NWAVES     = 8;
constexpr int ACT_TOT_H  = NWAVES * ACT_WAVE_H;
constexpr int BIAS_OFF_B = (WTOT_H + ACT_TOT_H) * 2;
constexpr int BIAS_F     = NLEV * 257;
constexpr size_t SMEM_BYTES = ((BIAS_OFF_B + BIAS_F * 4) + 15) & ~size_t(15); // ~321.8KB
static_assert(SMEM_BYTES <= 320u * 1024u, "exceeds CDNA5 LDS");

struct Params {
  const float* x;
  const float* fm[NLEV];           // original [C,H,W] f32
  const _Float16* fmT[NLEV];       // transposed [H*W,C] f16 (workspace)
  const float* W0[NLEV]; const float* B0[NLEV];
  const float* W1[NLEV]; const float* B1[NLEV];
  const float* W2[NLEV]; const float* B2[NLEV];
  float* out;
  int n, npairs;
};

__device__ __forceinline__ int widx(int S, int N, int k, int n) {
  return (((((k & 31) >> 3) * S) + (k >> 5)) * N + n) * 8 + (k & 7);
}

__device__ __forceinline__ void wsync() {
  __builtin_amdgcn_wave_barrier();
  asm volatile("s_wait_dscnt 0" ::: "memory");
  __builtin_amdgcn_wave_barrier();
}

__device__ __forceinline__ v16h cat8(v8h lo, v8h hi) {
  return __builtin_shufflevector(lo, hi, 0,1,2,3,4,5,6,7,8,9,10,11,12,13,14,15);
}

__device__ __forceinline__ v16h load_bfrag(const _Float16* wl, int baseh,
                                           int S, int N, int s, int nn, int hs) {
  v8h lo = *(const v8h*)(wl + baseh + (((hs    ) * S + s) * N + nn) * 8);
  v8h hi = *(const v8h*)(wl + baseh + (((2 + hs) * S + s) * N + nn) * 8);
  return cat8(lo, hi);
}

__device__ __forceinline__ v16h load_afrag(const _Float16* row, int s, int hs) {
  v8h lo = *(const v8h*)(row + s * 32 + hs * 8);
  v8h hi = *(const v8h*)(row + s * 32 + hs * 8 + 16);
  return cat8(lo, hi);
}

__device__ __forceinline__ v8f wmma16(v16h a, v16h b, v8f c) {
  return __builtin_amdgcn_wmma_f32_16x16x32_f16(false, a, false, b,
                                                (short)0, c, false, false);
}

__device__ __forceinline__ int iclamp(int v, int lo, int hi) {
  return v < lo ? lo : (v > hi ? hi : v);
}

// ------------- feature-map transpose: [C,H,W] f32 -> [H*W,C] f16 -----------
struct TP { const float* src[NLEV]; _Float16* dst[NLEV]; };

__global__ __launch_bounds__(256)
void fm_transpose(TP T) {
  const int lev = blockIdx.y;
  const int dim = (4 << lev) + 1;
  const int chw = dim * dim;
  const int tot = 32 * chw;
  for (int i = blockIdx.x * 256 + threadIdx.x; i < tot; i += gridDim.x * 256) {
    const int c = i & 31, hw = i >> 5;
    T.dst[lev][i] = (_Float16)T.src[lev][c * chw + hw];
  }
}

template <bool USET>
__global__ __launch_bounds__(256, 1)
void fused_grid_mlp(Params P) {
  extern __shared__ char smem[];
  _Float16* wl  = (_Float16*)smem;
  _Float16* act = wl + WTOT_H;
  float*    bl  = (float*)(smem + BIAS_OFF_B);

  const int tid = threadIdx.x;

  // ---------------- cooperative weight preload (f32 -> f16, swizzled) ----
  for (int lev = 0; lev < NLEV; ++lev) {
    const int wb = lev * LVL_H;
    const float* w0 = P.W0[lev];
    for (int i = tid; i < 64 * 128; i += 256) {
      int k = i >> 7, nn = i & 127;
      float v = (k < 34) ? w0[k * HID + nn] : 0.f;
      wl[wb + widx(2, 128, k, nn)] = (_Float16)v;
    }
    const float* w1 = P.W1[lev];
    for (int i = tid; i < 128 * 128; i += 256) {
      int k = i >> 7, nn = i & 127;
      wl[wb + W0_H + widx(4, 128, k, nn)] = (_Float16)w1[i];
    }
    const float* w2 = P.W2[lev];
    for (int i = tid; i < 128; i += 256) {
      int k = i;
      wl[wb + W0_H + W1_H + ((((k & 31) >> 3) * 4) + (k >> 5)) * 8 + (k & 7)]
          = (_Float16)w2[k];
    }
    const float* b0 = P.B0[lev]; const float* b1 = P.B1[lev];
    for (int i = tid; i < 128; i += 256) {
      bl[lev * 257 + i]       = b0[i];
      bl[lev * 257 + 128 + i] = b1[i];
    }
    if (tid == 0) bl[lev * 257 + 256] = P.B2[lev][0];
  }
  __syncthreads();

  const int lane = tid & 31;
  const int wv   = tid >> 5;
  const int hs   = lane >> 4;
  const int l16  = lane & 15;
  _Float16* abase = act + wv * ACT_WAVE_H;
  _Float16* rrow[2] = { abase + l16 * ACT_ROW_H,
                        abase + ACT_TILE_H + l16 * ACT_ROW_H };
  _Float16* wrow = abase + hs * ACT_TILE_H + l16 * ACT_ROW_H;

  // zero K pad halves 34..63 once (W0 rows there are zero; this kills
  // uninitialized-LDS NaNs on the first pair)
  for (int k = 34; k < 64; ++k) wrow[k] = (_Float16)0.f;
  wsync();

  const int gw      = blockIdx.x * NWAVES + wv;
  const int wstride = gridDim.x * NWAVES;

  for (int pair = gw; pair < P.npairs; pair += wstride) {
    const int base = pair * 32;
    int p = base + lane; if (p >= P.n) p = P.n - 1;
    const float xc = P.x[2 * p + 0];
    const float yc = P.x[2 * p + 1];

    float accf[32];
    #pragma unroll
    for (int c = 0; c < 32; ++c) accf[c] = 0.f;

    #pragma unroll
    for (int lev = 0; lev < NLEV; ++lev) {
      const int dim = (4 << lev) + 1;
      const int chw = dim * dim;

      // ---- bilinear grid_sample (align_corners=False, zeros padding) ----
      float xs = (xc + 1.f) * dim * 0.5f - 0.5f;
      float ys = (yc + 1.f) * dim * 0.5f - 0.5f;
      float fx0 = floorf(xs), fy0 = floorf(ys);
      float tx = xs - fx0, ty = ys - fy0;
      int ix0 = (int)fx0, iy0 = (int)fy0;
      #pragma unroll
      for (int corner = 0; corner < 4; ++corner) {
        const int dx = corner & 1, dy = corner >> 1;
        int xcc = ix0 + dx, ycc = iy0 + dy;
        float w = (dx ? tx : 1.f - tx) * (dy ? ty : 1.f - ty);
        if (xcc < 0 || xcc >= dim || ycc < 0 || ycc >= dim) w = 0.f;
        int xi = iclamp(xcc, 0, dim - 1);
        int yi = iclamp(ycc, 0, dim - 1);
        if constexpr (USET) {
          // [H*W, 32] f16: one 64B contiguous burst -> 4x global_load_b128
          const _Float16* srcT = P.fmT[lev] + (size_t)(yi * dim + xi) * 32;
          #pragma unroll
          for (int j = 0; j < 4; ++j) {
            v8h ch = *(const v8h*)(srcT + j * 8);
            #pragma unroll
            for (int e = 0; e < 8; ++e)
              accf[j * 8 + e] = fmaf(w, (float)ch[e], accf[j * 8 + e]);
          }
        } else {
          const float* src = P.fm[lev] + (yi * dim + xi);
          #pragma unroll
          for (int c = 0; c < 32; ++c)
            accf[c] = fmaf(w, src[c * chw], accf[c]);
        }
      }

      // ---- write layer-0 input row: [x, y, acc(32), 0-pad] as f16 ----
      wrow[0] = (_Float16)xc; wrow[1] = (_Float16)yc;
      #pragma unroll
      for (int c = 0; c < 32; ++c) wrow[2 + c] = (_Float16)accf[c];
      wsync();

      const int wb = lev * LVL_H;
      const int bb = lev * 257;

      // ---- layer 0: 2x [16x64] @ [64x128], bias + relu ----
      v16h a[2][2];
      #pragma unroll
      for (int q = 0; q < 2; ++q) {
        a[q][0] = load_afrag(rrow[q], 0, hs);
        a[q][1] = load_afrag(rrow[q], 1, hs);
      }
      wsync();
      #pragma unroll
      for (int t = 0; t < 8; ++t) {
        const int nn = t * 16 + l16;
        v16h b0 = load_bfrag(wl, wb, 2, 128, 0, nn, hs);
        v16h b1 = load_bfrag(wl, wb, 2, 128, 1, nn, hs);
        v8f acc0 = {}, acc1 = {};
        acc0 = wmma16(a[0][0], b0, acc0);
        acc1 = wmma16(a[1][0], b0, acc1);
        acc0 = wmma16(a[0][1], b1, acc0);
        acc1 = wmma16(a[1][1], b1, acc1);
        const float bsv = bl[bb + nn];
        #pragma unroll
        for (int v = 0; v < 8; ++v) {
          float h0 = acc0[v] + bsv; h0 = h0 > 0.f ? h0 : 0.f;
          float h1 = acc1[v] + bsv; h1 = h1 > 0.f ? h1 : 0.f;
          abase[(hs * 8 + v) * ACT_ROW_H + nn]              = (_Float16)h0;
          abase[ACT_TILE_H + (hs * 8 + v) * ACT_ROW_H + nn] = (_Float16)h1;
        }
      }
      wsync();

      // ---- layer 1: 2x [16x128] @ [128x128], bias + relu ----
      v16h h[2][4];
      #pragma unroll
      for (int q = 0; q < 2; ++q)
        #pragma unroll
        for (int s = 0; s < 4; ++s) h[q][s] = load_afrag(rrow[q], s, hs);
      wsync();
      #pragma unroll
      for (int t = 0; t < 8; ++t) {
        const int nn = t * 16 + l16;
        v8f acc0 = {}, acc1 = {};
        #pragma unroll
        for (int s = 0; s < 4; ++s) {
          v16h b = load_bfrag(wl, wb + W0_H, 4, 128, s, nn, hs);
          acc0 = wmma16(h[0][s], b, acc0);
          acc1 = wmma16(h[1][s], b, acc1);
        }
        const float bsv = bl[bb + 128 + nn];
        #pragma unroll
        for (int v = 0; v < 8; ++v) {
          float h0 = acc0[v] + bsv; h0 = h0 > 0.f ? h0 : 0.f;
          float h1 = acc1[v] + bsv; h1 = h1 > 0.f ? h1 : 0.f;
          abase[(hs * 8 + v) * ACT_ROW_H + nn]              = (_Float16)h0;
          abase[ACT_TILE_H + (hs * 8 + v) * ACT_ROW_H + nn] = (_Float16)h1;
        }
      }
      wsync();

      // ---- layer 2: 2x [16x128] @ [128x16]; only column 0 is real ----
      v16h g[2][4];
      #pragma unroll
      for (int q = 0; q < 2; ++q)
        #pragma unroll
        for (int s = 0; s < 4; ++s) g[q][s] = load_afrag(rrow[q], s, hs);
      const int wb2 = wb + W0_H + W1_H;
      v16h w2f[4];
      #pragma unroll
      for (int s = 0; s < 4; ++s) {
        v8h lo = {}, hi = {};
        if (l16 == 0) {
          lo = *(const v8h*)(wl + wb2 + ((hs    ) * 4 + s) * 8);
          hi = *(const v8h*)(wl + wb2 + ((2 + hs) * 4 + s) * 8);
        }
        w2f[s] = cat8(lo, hi);
      }
      v8f o0 = {}, o1 = {};
      #pragma unroll
      for (int s = 0; s < 4; ++s) {
        o0 = wmma16(g[0][s], w2f[s], o0);
        o1 = wmma16(g[1][s], w2f[s], o1);
      }
      if (l16 == 0) {
        const float b2v = bl[bb + 256];
        #pragma unroll
        for (int v = 0; v < 8; ++v) {
          const int pr0 = base + hs * 8 + v;
          const int pr1 = base + 16 + hs * 8 + v;
          if (pr0 < P.n) P.out[(size_t)lev * (size_t)P.n + pr0] = o0[v] + b2v;
          if (pr1 < P.n) P.out[(size_t)lev * (size_t)P.n + pr1] = o1[v] + b2v;
        }
      }
      wsync();
    }
  }
}

extern "C" void kernel_launch(void* const* d_in, const int* in_sizes, int n_in,
                              void* d_out, int out_size, void* d_ws, size_t ws_size,
                              hipStream_t stream) {
  (void)n_in; (void)out_size;
  Params P;
  P.x = (const float*)d_in[0];
  for (int l = 0; l < NLEV; ++l) P.fm[l] = (const float*)d_in[1 + l];
  for (int l = 0; l < NLEV; ++l) {
    const int b = 6 + l * 6;   // per level: W0, b0, W1, b1, W2, b2
    P.W0[l] = (const float*)d_in[b + 0]; P.B0[l] = (const float*)d_in[b + 1];
    P.W1[l] = (const float*)d_in[b + 2]; P.B1[l] = (const float*)d_in[b + 3];
    P.W2[l] = (const float*)d_in[b + 4]; P.B2[l] = (const float*)d_in[b + 5];
  }
  P.out    = (float*)d_out;
  P.n      = in_sizes[0] / 2;
  P.npairs = (P.n + 31) / 32;

  // workspace layout for transposed f16 feature pyramid
  size_t off = 0;
  TP T;
  for (int l = 0; l < NLEV; ++l) {
    const int dim = (4 << l) + 1;
    T.src[l]  = P.fm[l];
    T.dst[l]  = (_Float16*)d_ws + off;
    P.fmT[l]  = T.dst[l];
    off += (size_t)32 * dim * dim;
  }
  const bool useT = ws_size >= off * sizeof(_Float16);

  if (useT) {
    // largest level: 32*65*65 = 135200 elems -> 529 blocks covers it
    fm_transpose<<<dim3(529, NLEV), 256, 0, stream>>>(T);
    (void)hipFuncSetAttribute((const void*)fused_grid_mlp<true>,
                              hipFuncAttributeMaxDynamicSharedMemorySize,
                              (int)SMEM_BYTES);
    fused_grid_mlp<true><<<1024, 256, SMEM_BYTES, stream>>>(P);
  } else {
    for (int l = 0; l < NLEV; ++l) P.fmT[l] = nullptr;
    (void)hipFuncSetAttribute((const void*)fused_grid_mlp<false>,
                              hipFuncAttributeMaxDynamicSharedMemorySize,
                              (int)SMEM_BYTES);
    fused_grid_mlp<false><<<1024, 256, SMEM_BYTES, stream>>>(P);
  }
}
